// Embed_tft_58317065945189
// MI455X (gfx1250) — matrix operand order repaired and checked by reading, not compiled, not run on hardware
//
#include <hip/hip_runtime.h>

#define B_   32
#define T_   512
#define LAG_ 96
#define TP_  416   // T - LAG
#define E_   128
#define V_   512
#define NF_  8
#define OFF_DEC (B_*TP_*E_)   // start of output_dec in d_out

typedef __bf16 bf16_t;
typedef __attribute__((ext_vector_type(16))) __bf16 v16bf;
typedef __attribute__((ext_vector_type(8)))  __bf16 v8bf;
typedef __attribute__((ext_vector_type(8)))  float  v8f;

// ---------------------------------------------------------------- helpers

__device__ __forceinline__ v8f wmma_bf16(v16bf a, v16bf b, v8f c) {
  // D = A(16x32 bf16) x B(32x16 bf16) + C(16x16 f32)
  return __builtin_amdgcn_wmma_f32_16x16x32_bf16(false, a, false, b,
                                                 (short)0, c, false, false);
}

__device__ __forceinline__ v16bf make_v16(v8bf lo, v8bf hi) {
  v16bf r;
#pragma unroll
  for (int i = 0; i < 8; ++i) { r[i] = lo[i]; r[i + 8] = hi[i]; }
  return r;
}

// A fragment (16x32 bf16) from a row-major LDS tile.
// Layout used here (B swizzle matches): lane L holds row L&15; half=L>>4 selects K-halves:
//   elems 0..15 -> K = kt*32 + half*16 + e
// Contiguous run -> two ds_load_b128 per fragment.
__device__ __forceinline__ v16bf load_a_frag(const bf16_t* t, int stride,
                                             int kt, int lane) {
  int row = lane & 15, hf = lane >> 4;
  const bf16_t* p = t + row * stride + kt * 32 + hf * 16;
  v8bf lo = *(const v8bf*)p;
  v8bf hi = *(const v8bf*)(p + 8);
  return make_v16(lo, hi);
}

// B fragment (32x16 bf16), preswizzled: per (kt,nt,lane) 16 contiguous bf16
// (elem e -> K = kt*32 + half*16 + e, N = nt*16 + (lane&15)).
__device__ __forceinline__ v16bf load_b_frag(const bf16_t* swz, int NT,
                                             int kt, int nt, int lane) {
  const bf16_t* p = swz + (((kt * NT + nt) * 32 + lane) << 4);
  v8bf lo = *(const v8bf*)p;
  v8bf hi = *(const v8bf*)(p + 8);
  return make_v16(lo, hi);
}

__device__ __forceinline__ float sig_(float x)  { return 1.0f / (1.0f + __expf(-x)); }
__device__ __forceinline__ float elu_(float x)  { return x > 0.f ? x : __expf(x) - 1.f; }
__device__ __forceinline__ float tanh_(float x) {
  float ax = fabsf(x);
  float e  = __expf(-2.f * ax);
  float t  = (1.f - e) / (1.f + e);
  return x < 0.f ? -t : t;
}

// -------------------------------------------------------- weight swizzling
// Pre-swizzle W[o][i] row-major f32 into bf16 B-fragments: out = x @ W^T,
// i.e. B(K=i, N=o) = W[N][K].

__global__ void swizzle_grn_kernel(bf16_t* __restrict__ dst,
                                   const float* __restrict__ w1,
                                   const float* __restrict__ w2,
                                   const float* __restrict__ gw1,
                                   const float* __restrict__ gw2) {
  int mat = blockIdx.x;            // f*4 + m, 32 mats of 128x128
  int f = mat >> 2, m = mat & 3;
  const float* src = (m == 0 ? w1 : m == 1 ? w2 : m == 2 ? gw1 : gw2)
                     + (size_t)f * E_ * E_;
  bf16_t* d = dst + (size_t)mat * 16384;
  for (int i = threadIdx.x; i < 16384; i += blockDim.x) {
    int e = i & 15, lane = (i >> 4) & 31, fr = i >> 9;
    int nt = fr & 7, kt = fr >> 3;
    int hf = lane >> 4;
    int K = kt * 32 + hf * 16 + e;
    int N = nt * 16 + (lane & 15);
    d[i] = (bf16_t)src[N * E_ + K];
  }
}

__global__ void swizzle_lstm_kernel(bf16_t* __restrict__ dst,
                                    const float* __restrict__ wih,
                                    const float* __restrict__ whh) {
  int l = blockIdx.x;              // layer, K=256 (x|h), N=512
  const float* wi = wih + (size_t)l * 512 * E_;
  const float* wh = whh + (size_t)l * 512 * E_;
  bf16_t* d = dst + (size_t)l * 131072;
  for (int i = threadIdx.x; i < 131072; i += blockDim.x) {
    int e = i & 15, lane = (i >> 4) & 31, fr = i >> 9;
    int nt = fr & 31, kt = fr >> 5;
    int hf = lane >> 4;
    int K = kt * 32 + hf * 16 + e;
    int N = nt * 16 + (lane & 15);
    float v = (K < 128) ? wi[N * E_ + K] : wh[N * E_ + (K - 128)];
    d[i] = (bf16_t)v;
  }
}

__global__ void swizzle_glu_kernel(bf16_t* __restrict__ dst,
                                   const float* __restrict__ a,
                                   const float* __restrict__ b,
                                   const float* __restrict__ c,
                                   const float* __restrict__ dd) {
  int mat = blockIdx.x;            // 0..3: enc_gw1, enc_gw2, dec_gw1, dec_gw2
  const float* src = mat == 0 ? a : mat == 1 ? b : mat == 2 ? c : dd;
  bf16_t* d = dst + (size_t)mat * 16384;
  for (int i = threadIdx.x; i < 16384; i += blockDim.x) {
    int e = i & 15, lane = (i >> 4) & 31, fr = i >> 9;
    int nt = fr & 7, kt = fr >> 3;
    int hf = lane >> 4;
    int K = kt * 32 + hf * 16 + e;
    int N = nt * 16 + (lane & 15);
    d[i] = (bf16_t)src[N * E_ + K];
  }
}

// ------------------------------------------------- embed + GRN + VSN combine
// One wave per 16-token tile. Barrier-free & deterministic: the 9-way
// weighted combine is accumulated in registers in fixed feature order.

__global__ __launch_bounds__(32) void embed_vsn_kernel(
    const int*   __restrict__ xin, const float* __restrict__ y,
    const float* __restrict__ emb_ne,
    const float* __restrict__ gne_b1, const float* __restrict__ gne_b2,
    const float* __restrict__ gne_gb1, const float* __restrict__ gne_gb2,
    const float* __restrict__ gne_lng, const float* __restrict__ gne_lnb,
    const float* __restrict__ g1_lnb,  const float* __restrict__ yw_ne,
    const float* __restrict__ yw_1,
    const bf16_t* __restrict__ swz_grn,
    float* __restrict__ emb_out) {
  __shared__ __align__(16) bf16_t bufA[16 * 136];
  __shared__ __align__(16) bf16_t bufB[16 * 136];
  __shared__ float wsm[16 * 10];
  __shared__ float yrow[16];

  const int tile = blockIdx.x;            // 1024 tiles (T/16=32 per batch)
  const int b = tile >> 5;
  const int t0 = (tile & 31) * 16;
  const bool isFut = (t0 >= TP_);         // TP_ is 16-aligned: no mixed tiles
  const int lane = threadIdx.x, hf = lane >> 4, lcol = lane & 15;

  // Per-row softmax weights over 9 logits (8 constants g1_lnb[f] + y*yw_1).
  if (lane < 16) {
    int t = t0 + lane;
    int ty = isFut ? t - 1 : t;
    float yv = y[b * T_ + ty];
    yrow[lane] = yv;
    float l8 = yv * yw_1[0];
    float lf[8], mx = l8;
#pragma unroll
    for (int f = 0; f < NF_; ++f) { lf[f] = g1_lnb[f]; mx = fmaxf(mx, lf[f]); }
    float s = 0.f, ef[8];
#pragma unroll
    for (int f = 0; f < NF_; ++f) { ef[f] = __expf(lf[f] - mx); s += ef[f]; }
    float e8 = __expf(l8 - mx); s += e8;
    float inv = 1.f / s;
#pragma unroll
    for (int f = 0; f < NF_; ++f) wsm[lane * 10 + f] = ef[f] * inv;
    wsm[lane * 10 + 8] = e8 * inv;
  }

  // Register accumulator (D-frag layout), init with ey_ne * w8 / 9.
  float acc64[64];
#pragma unroll
  for (int nt = 0; nt < 8; ++nt)
#pragma unroll
    for (int v = 0; v < 8; ++v) {
      int row = v + 8 * hf, col = nt * 16 + lcol;
      acc64[nt * 8 + v] =
          yrow[row] * yw_ne[col] * wsm[row * 10 + 8] * (1.f / 9.f);
    }

  for (int f = 0; f < NF_; ++f) {
    // ---- gather embedding rows into bufA (bf16, A layout source) ----
    {
      int r = lane >> 1;
      int t = t0 + r;
      int idx;
      if (f < 5)       idx = xin[(b * T_ + t) * 6 + (f + 1)];
      else if (f == 5) idx = t;
      else if (f == 6) idx = (t >= TP_) ? (t - TP_ + 1) : 0;
      else             idx = (t >= TP_) ? 1 : 0;
      const float* src = emb_ne + ((size_t)f * V_ + idx) * E_ + (lane & 1) * 64;
      bf16_t* dst = &bufA[r * 136 + (lane & 1) * 64];
#pragma unroll
      for (int j = 0; j < 16; ++j) {
        float4 v = *(const float4*)(src + j * 4);
        dst[j * 4 + 0] = (bf16_t)v.x; dst[j * 4 + 1] = (bf16_t)v.y;
        dst[j * 4 + 2] = (bf16_t)v.z; dst[j * 4 + 3] = (bf16_t)v.w;
      }
    }

    const bf16_t* wz1 = swz_grn + (size_t)(f * 4 + 0) * 16384;
    const bf16_t* wz2 = swz_grn + (size_t)(f * 4 + 1) * 16384;
    const bf16_t* wzs = swz_grn + (size_t)(f * 4 + 2) * 16384;
    const bf16_t* wzl = swz_grn + (size_t)(f * 4 + 3) * 16384;

    // ---- matmul 1: eta1 = elu(x @ w1^T + b1) -> bufB ----
#pragma unroll
    for (int nt = 0; nt < 8; ++nt) {
      float bb = gne_b1[f * E_ + nt * 16 + lcol];
      v8f acc;
#pragma unroll
      for (int i = 0; i < 8; ++i) acc[i] = bb;
#pragma unroll
      for (int kt = 0; kt < 4; ++kt) {
        v16bf a = load_a_frag(bufA, 136, kt, lane);
        acc = wmma_bf16(a, load_b_frag(wz1, 8, kt, nt, lane), acc);
      }
#pragma unroll
      for (int v = 0; v < 8; ++v)
        bufB[(v + 8 * hf) * 136 + nt * 16 + lcol] = (bf16_t)elu_(acc[v]);
    }

    // residual x in D layout (before bufA is reused for eta2)
    float xres[64];
#pragma unroll
    for (int nt = 0; nt < 8; ++nt)
#pragma unroll
      for (int v = 0; v < 8; ++v)
        xres[nt * 8 + v] = (float)bufA[(v + 8 * hf) * 136 + nt * 16 + lcol];

    // ---- matmul 2: eta2 = eta1 @ w2^T + b2 -> bufA ----
#pragma unroll
    for (int nt = 0; nt < 8; ++nt) {
      float bb = gne_b2[f * E_ + nt * 16 + lcol];
      v8f acc;
#pragma unroll
      for (int i = 0; i < 8; ++i) acc[i] = bb;
#pragma unroll
      for (int kt = 0; kt < 4; ++kt) {
        v16bf a = load_a_frag(bufB, 136, kt, lane);
        acc = wmma_bf16(a, load_b_frag(wz2, 8, kt, nt, lane), acc);
      }
#pragma unroll
      for (int v = 0; v < 8; ++v)
        bufA[(v + 8 * hf) * 136 + nt * 16 + lcol] = (bf16_t)acc[v];
    }

    // ---- GLU: sigmoid(eta2@gw1+gb1) * (eta2@gw2+gb2) + x ----
    float res[64];
#pragma unroll
    for (int nt = 0; nt < 8; ++nt) {
      float bs = gne_gb1[f * E_ + nt * 16 + lcol];
      float bl = gne_gb2[f * E_ + nt * 16 + lcol];
      v8f as, al;
#pragma unroll
      for (int i = 0; i < 8; ++i) { as[i] = bs; al[i] = bl; }
#pragma unroll
      for (int kt = 0; kt < 4; ++kt) {
        v16bf a = load_a_frag(bufA, 136, kt, lane);
        as = wmma_bf16(a, load_b_frag(wzs, 8, kt, nt, lane), as);
        al = wmma_bf16(a, load_b_frag(wzl, 8, kt, nt, lane), al);
      }
#pragma unroll
      for (int v = 0; v < 8; ++v)
        res[nt * 8 + v] = sig_(as[v]) * al[v] + xres[nt * 8 + v];
    }

    // ---- LayerNorm over E via cross-lane shuffles (rows live in 16 lanes) --
    float mean[8], inv[8];
#pragma unroll
    for (int v = 0; v < 8; ++v) {
      float p = 0.f, q = 0.f;
#pragma unroll
      for (int nt = 0; nt < 8; ++nt) { float r = res[nt * 8 + v]; p += r; q += r * r; }
#pragma unroll
      for (int m = 1; m < 16; m <<= 1) {
        p += __shfl_xor(p, m, 32);
        q += __shfl_xor(q, m, 32);
      }
      float mu  = p * (1.f / 128.f);
      float var = fmaxf(q * (1.f / 128.f) - mu * mu, 0.f);
      mean[v] = mu;
      inv[v]  = rsqrtf(var + 1e-5f);
    }
    float wrow[8];
#pragma unroll
    for (int v = 0; v < 8; ++v)
      wrow[v] = wsm[(v + 8 * hf) * 10 + f] * (1.f / 9.f);
#pragma unroll
    for (int nt = 0; nt < 8; ++nt) {
      float g  = gne_lng[f * E_ + nt * 16 + lcol];
      float bb = gne_lnb[f * E_ + nt * 16 + lcol];
#pragma unroll
      for (int v = 0; v < 8; ++v)
        acc64[nt * 8 + v] +=
            ((res[nt * 8 + v] - mean[v]) * inv[v] * g + bb) * wrow[v];
    }
  }

  // write combined embedding (emb buffer indexed by absolute t)
#pragma unroll
  for (int nt = 0; nt < 8; ++nt)
#pragma unroll
    for (int v = 0; v < 8; ++v) {
      int row = v + 8 * hf;
      emb_out[((size_t)(b * T_ + t0 + row)) * E_ + nt * 16 + lcol] =
          acc64[nt * 8 + v];
    }
}

// ------------------------------------------------------------- LSTM scan
// grid = 4 blocks: (net enc/dec) x (batch half of 16 rows). Persistent
// sequential scan: h/c in LDS, gates via WMMA over K=256 [x|h] concat.

__global__ __launch_bounds__(256) void lstm_kernel(
    const float* __restrict__ emb, float* __restrict__ buf0,
    float* __restrict__ buf1,
    const bf16_t* __restrict__ swz_enc, const bf16_t* __restrict__ swz_dec,
    const float* __restrict__ enc_bih, const float* __restrict__ enc_bhh,
    const float* __restrict__ dec_bih, const float* __restrict__ dec_bhh) {
  __shared__ __align__(16) bf16_t xh[16 * 264];   // [x(128) | h(128)] bf16
  __shared__ float gates[16 * 520];
  __shared__ float cst[16 * 128];

  const int net = blockIdx.x >> 1;                 // 0 enc, 1 dec
  const int b0  = (blockIdx.x & 1) * 16;
  const int Tn  = net ? LAG_ : TP_;
  const int tb  = net ? TP_  : 0;
  const bf16_t* swz = net ? swz_dec : swz_enc;
  const float* bih = net ? dec_bih : enc_bih;
  const float* bhh = net ? dec_bhh : enc_bhh;

  const int tid = threadIdx.x, wave = tid >> 5, lane = tid & 31;
  const int hf = lane >> 4, lcol = lane & 15;

  for (int l = 0; l < 3; ++l) {
    const float* inbuf  = (l == 0) ? emb  : ((l == 1) ? buf0 : buf1);
    float*       outbuf = (l == 0) ? buf0 : ((l == 1) ? buf1 : buf0);
    const bf16_t* wl = swz + (size_t)l * 131072;

    for (int i = tid; i < 16 * 128; i += 256) {   // h = c = 0
      cst[i] = 0.f;
      xh[(i >> 7) * 264 + 128 + (i & 127)] = (bf16_t)0.f;
    }
    __syncthreads();

    for (int t = 0; t < Tn; ++t) {
      for (int i = tid; i < 16 * 128; i += 256) { // stage x_t as bf16
        int r = i >> 7, c = i & 127;
        xh[r * 264 + c] =
            (bf16_t)inbuf[((size_t)(b0 + r) * T_ + tb + t) * E_ + c];
      }
      __syncthreads();

      // gates(16x512) = [x|h](16x256) @ W^T + bih + bhh
      v8f acc[4];
#pragma unroll
      for (int j = 0; j < 4; ++j) {
        int nt = wave * 4 + j;
        float bb = bih[l * 512 + nt * 16 + lcol] + bhh[l * 512 + nt * 16 + lcol];
#pragma unroll
        for (int i = 0; i < 8; ++i) acc[j][i] = bb;
      }
#pragma unroll
      for (int kt = 0; kt < 8; ++kt) {
        v16bf a = load_a_frag(xh, 264, kt, lane);
        if (kt < 7)
          __builtin_prefetch(wl + ((((kt + 1) * 32 + wave * 4) * 32 + lane) << 4), 0, 0);
#pragma unroll
        for (int j = 0; j < 4; ++j)
          acc[j] = wmma_bf16(a, load_b_frag(wl, 32, kt, wave * 4 + j, lane), acc[j]);
      }
#pragma unroll
      for (int j = 0; j < 4; ++j) {
        int nt = wave * 4 + j;
#pragma unroll
        for (int v = 0; v < 8; ++v)
          gates[(v + 8 * hf) * 520 + nt * 16 + lcol] = acc[j][v];
      }
      __syncthreads();

      // cell update
      for (int i = tid; i < 16 * 128; i += 256) {
        int r = i >> 7, j = i & 127;
        float gi = gates[r * 520 + j];
        float gf = gates[r * 520 + 128 + j];
        float gg = gates[r * 520 + 256 + j];
        float go = gates[r * 520 + 384 + j];
        float cn = sig_(gf) * cst[r * 128 + j] + sig_(gi) * tanh_(gg);
        float h  = sig_(go) * tanh_(cn);
        cst[r * 128 + j] = cn;
        xh[r * 264 + 128 + j] = (bf16_t)h;
        outbuf[((size_t)(b0 + r) * T_ + tb + t) * E_ + j] = h;
      }
      __syncthreads();
    }
    __syncthreads();
  }
}

// ------------------------------------------------ GLU + residual + LN head
// One wave per 16-token tile; final LSTM output is in buf0.

__global__ __launch_bounds__(32) void glu_out_kernel(
    const float* __restrict__ h_in, const float* __restrict__ emb,
    const bf16_t* __restrict__ swz_glu,
    const float* __restrict__ enc_gb1, const float* __restrict__ enc_gb2,
    const float* __restrict__ enc_lng, const float* __restrict__ enc_lnb,
    const float* __restrict__ dec_gb1, const float* __restrict__ dec_gb2,
    const float* __restrict__ dec_lng, const float* __restrict__ dec_lnb,
    float* __restrict__ out) {
  __shared__ __align__(16) bf16_t hA[16 * 136];

  const int tile = blockIdx.x;
  const int b = tile >> 5;
  const int t0 = (tile & 31) * 16;
  const int net = (t0 >= TP_) ? 1 : 0;
  const int lane = threadIdx.x, hf = lane >> 4, lcol = lane & 15;

  const bf16_t* wzs = swz_glu + (size_t)(net ? 2 : 0) * 16384;
  const bf16_t* wzl = wzs + 16384;
  const float* gb1 = net ? dec_gb1 : enc_gb1;
  const float* gb2 = net ? dec_gb2 : enc_gb2;
  const float* lng = net ? dec_lng : enc_lng;
  const float* lnb = net ? dec_lnb : enc_lnb;

  {  // stage h tile as bf16
    int r = lane >> 1;
    const float* src = h_in + ((size_t)(b * T_ + t0 + r)) * E_ + (lane & 1) * 64;
    bf16_t* dst = &hA[r * 136 + (lane & 1) * 64];
#pragma unroll
    for (int j = 0; j < 16; ++j) {
      float4 v = *(const float4*)(src + j * 4);
      dst[j * 4 + 0] = (bf16_t)v.x; dst[j * 4 + 1] = (bf16_t)v.y;
      dst[j * 4 + 2] = (bf16_t)v.z; dst[j * 4 + 3] = (bf16_t)v.w;
    }
  }

  float res[64];
#pragma unroll
  for (int nt = 0; nt < 8; ++nt) {
    float bs = gb1[nt * 16 + lcol], bl = gb2[nt * 16 + lcol];
    v8f as, al;
#pragma unroll
    for (int i = 0; i < 8; ++i) { as[i] = bs; al[i] = bl; }
#pragma unroll
    for (int kt = 0; kt < 4; ++kt) {
      v16bf a = load_a_frag(hA, 136, kt, lane);
      as = wmma_bf16(a, load_b_frag(wzs, 8, kt, nt, lane), as);
      al = wmma_bf16(a, load_b_frag(wzl, 8, kt, nt, lane), al);
    }
#pragma unroll
    for (int v = 0; v < 8; ++v) {
      int row = v + 8 * hf;
      float r0 = emb[((size_t)(b * T_ + t0 + row)) * E_ + nt * 16 + lcol];
      res[nt * 8 + v] = sig_(as[v]) * al[v] + r0;
    }
  }

  float mean[8], inv[8];
#pragma unroll
  for (int v = 0; v < 8; ++v) {
    float p = 0.f, q = 0.f;
#pragma unroll
    for (int nt = 0; nt < 8; ++nt) { float r = res[nt * 8 + v]; p += r; q += r * r; }
#pragma unroll
    for (int m = 1; m < 16; m <<= 1) {
      p += __shfl_xor(p, m, 32);
      q += __shfl_xor(q, m, 32);
    }
    float mu  = p * (1.f / 128.f);
    float var = fmaxf(q * (1.f / 128.f) - mu * mu, 0.f);
    mean[v] = mu;
    inv[v]  = rsqrtf(var + 1e-5f);
  }

#pragma unroll
  for (int nt = 0; nt < 8; ++nt) {
    float g  = lng[nt * 16 + lcol];
    float bb = lnb[nt * 16 + lcol];
#pragma unroll
    for (int v = 0; v < 8; ++v) {
      int row = v + 8 * hf;
      int t = t0 + row;
      float val = (res[nt * 8 + v] - mean[v]) * inv[v] * g + bb;
      size_t o = net
          ? (size_t)OFF_DEC + ((size_t)b * LAG_ + (t - TP_)) * E_
          : ((size_t)b * TP_ + t) * E_;
      out[o + nt * 16 + lcol] = val;
    }
  }
}

// ----------------------------------------------------------------- launch

extern "C" void kernel_launch(void* const* d_in, const int* in_sizes, int n_in,
                              void* d_out, int out_size, void* d_ws,
                              size_t ws_size, hipStream_t stream) {
  const int*   x       = (const int*)  d_in[0];
  const float* y       = (const float*)d_in[1];
  const float* emb_ne  = (const float*)d_in[2];
  const float* gne_w1  = (const float*)d_in[4];
  const float* gne_b1  = (const float*)d_in[5];
  const float* gne_w2  = (const float*)d_in[6];
  const float* gne_b2  = (const float*)d_in[7];
  const float* gne_gw1 = (const float*)d_in[8];
  const float* gne_gb1 = (const float*)d_in[9];
  const float* gne_gw2 = (const float*)d_in[10];
  const float* gne_gb2 = (const float*)d_in[11];
  const float* gne_lng = (const float*)d_in[12];
  const float* gne_lnb = (const float*)d_in[13];
  const float* g1_lnb  = (const float*)d_in[23];
  const float* yw_ne   = (const float*)d_in[24];
  const float* yw_1    = (const float*)d_in[25];
  const float* enc_wih = (const float*)d_in[26];
  const float* enc_whh = (const float*)d_in[27];
  const float* enc_bih = (const float*)d_in[28];
  const float* enc_bhh = (const float*)d_in[29];
  const float* enc_gw1 = (const float*)d_in[30];
  const float* enc_gb1 = (const float*)d_in[31];
  const float* enc_gw2 = (const float*)d_in[32];
  const float* enc_gb2 = (const float*)d_in[33];
  const float* enc_lng = (const float*)d_in[34];
  const float* enc_lnb = (const float*)d_in[35];
  const float* dec_wih = (const float*)d_in[36];
  const float* dec_whh = (const float*)d_in[37];
  const float* dec_bih = (const float*)d_in[38];
  const float* dec_bhh = (const float*)d_in[39];
  const float* dec_gw1 = (const float*)d_in[40];
  const float* dec_gb1 = (const float*)d_in[41];
  const float* dec_gw2 = (const float*)d_in[42];
  const float* dec_gb2 = (const float*)d_in[43];
  const float* dec_lng = (const float*)d_in[44];
  const float* dec_lnb = (const float*)d_in[45];

  char* ws = (char*)d_ws;
  const size_t SZ_SEQ = (size_t)B_ * T_ * E_ * sizeof(float);   // 8 MB
  float*  emb      = (float*)(ws);
  float*  buf0     = (float*)(ws + SZ_SEQ);
  float*  buf1     = (float*)(ws + 2 * SZ_SEQ);
  bf16_t* swz_grn  = (bf16_t*)(ws + 3 * SZ_SEQ);                          // 1 MB
  bf16_t* swz_lenc = (bf16_t*)(ws + 3 * SZ_SEQ + 1048576);                // 768 KB
  bf16_t* swz_ldec = (bf16_t*)(ws + 3 * SZ_SEQ + 1048576 + 786432);       // 768 KB
  bf16_t* swz_glu  = (bf16_t*)(ws + 3 * SZ_SEQ + 1048576 + 2 * 786432);   // 128 KB

  swizzle_grn_kernel <<<32, 256, 0, stream>>>(swz_grn, gne_w1, gne_w2,
                                              gne_gw1, gne_gw2);
  swizzle_lstm_kernel<<<3, 256, 0, stream>>>(swz_lenc, enc_wih, enc_whh);
  swizzle_lstm_kernel<<<3, 256, 0, stream>>>(swz_ldec, dec_wih, dec_whh);
  swizzle_glu_kernel <<<4, 256, 0, stream>>>(swz_glu, enc_gw1, enc_gw2,
                                             dec_gw1, dec_gw2);

  embed_vsn_kernel<<<(B_ * T_) / 16, 32, 0, stream>>>(
      x, y, emb_ne, gne_b1, gne_b2, gne_gb1, gne_gb2, gne_lng, gne_lnb,
      g1_lnb, yw_ne, yw_1, swz_grn, emb);

  lstm_kernel<<<4, 256, 0, stream>>>(emb, buf0, buf1, swz_lenc, swz_ldec,
                                     enc_bih, enc_bhh, dec_bih, dec_bhh);

  glu_out_kernel<<<(B_ * T_) / 16 / 8 * 8 / 8, 256 / 8 * 0 + 32, 0, stream>>>(
      buf0, emb, swz_glu, enc_gb1, enc_gb2, enc_lng, enc_lnb, dec_gb1,
      dec_gb2, dec_lng, dec_lnb, (float*)d_out);
  // note: grid above simplifies to 128 blocks of 32 threads
  (void)in_sizes; (void)n_in; (void)out_size; (void)ws_size;
}
